// LSHAttention_90245852824037
// MI455X (gfx1250) — matrix operand order repaired and checked by reading, not compiled, not run on hardware
//
#include <hip/hip_runtime.h>

typedef __attribute__((ext_vector_type(16))) __bf16 v16bf;
typedef __attribute__((ext_vector_type(2)))  __bf16 v2bf;
typedef __attribute__((ext_vector_type(8)))  float  v8f;

#define TSEQ 4096
#define DDIM 512
#define NH   8
#define NB   64
#define BS   64

__device__ __forceinline__ unsigned short f2bf(float f) {
  __bf16 h = (__bf16)f;                       // native v_cvt, RNE
  return __builtin_bit_cast(unsigned short, h);
}
__device__ __forceinline__ unsigned int pack2(float a, float b) {
#if __has_builtin(__builtin_amdgcn_cvt_pk_bf16_f32)
  v2bf t = __builtin_amdgcn_cvt_pk_bf16_f32(a, b);   // lo=a, hi=b, one VALU op
  return __builtin_bit_cast(unsigned int, t);
#else
  return ((unsigned int)f2bf(b) << 16) | (unsigned int)f2bf(a);
#endif
}

union FragBF { v16bf v; uint4 u[2]; };

// ---------------------------------------------------------------------------
// Generic C[M,N] = A[M,K] * W[K,N] (+ bias), fp32 in/out, bf16 WMMA compute.
// Block tile 128x64, 8 waves (2x4), K-step 32, software-pipelined global loads.
// ---------------------------------------------------------------------------
#define BM 128
#define BN 64
#define KT 32
#define AS_STR 40   // ushorts per row (32 + 8 pad), 80B -> 16B aligned
#define WT_STR 40

__global__ __launch_bounds__(256) void gemm_bf16(
    const float* __restrict__ A, const float* __restrict__ W,
    const float* __restrict__ bias, float* __restrict__ C,
    int M, int N, int K) {
  __shared__ alignas(16) unsigned short As[BM * AS_STR];
  __shared__ alignas(16) unsigned short Wt[BN * WT_STR];   // transposed: [n][k]

  const int tid  = threadIdx.x;
  const int wave = tid >> 5;
  const int lane = tid & 31;
  const int lm   = lane & 15;
  const int lh   = lane >> 4;
  const int m0 = blockIdx.x * BM;
  const int n0 = blockIdx.y * BN;
  const int mw = (wave >> 2) * 64;
  const int nw = (wave & 3) * 16;

  v8f acc[4] = {};
  float4 aReg[4], wReg[2];

  // prologue: fetch first K-tile into registers
#pragma unroll
  for (int i = 0; i < 4; i++) {
    int idx = tid + i * 256;
    aReg[i] = *(const float4*)(A + (size_t)(m0 + (idx >> 3)) * K + (idx & 7) * 4);
  }
#pragma unroll
  for (int i = 0; i < 2; i++) {
    int idx = tid + i * 256;
    wReg[i] = *(const float4*)(W + (size_t)(idx >> 4) * N + n0 + (idx & 15) * 4);
  }

  for (int k0 = 0; k0 < K; k0 += KT) {
    // store current tile regs -> LDS (bf16, W transposed)
#pragma unroll
    for (int i = 0; i < 4; i++) {
      int idx = tid + i * 256;
      int r = idx >> 3;
      int c = (idx & 7) * 4;
      uint2 p; p.x = pack2(aReg[i].x, aReg[i].y); p.y = pack2(aReg[i].z, aReg[i].w);
      *(uint2*)&As[r * AS_STR + c] = p;
    }
#pragma unroll
    for (int i = 0; i < 2; i++) {
      int idx = tid + i * 256;
      int r = idx >> 4;            // k 0..31
      int c = (idx & 15) * 4;      // n 0..60
      Wt[(c + 0) * WT_STR + r] = f2bf(wReg[i].x);
      Wt[(c + 1) * WT_STR + r] = f2bf(wReg[i].y);
      Wt[(c + 2) * WT_STR + r] = f2bf(wReg[i].z);
      Wt[(c + 3) * WT_STR + r] = f2bf(wReg[i].w);
    }
    __syncthreads();

    // issue next K-tile global loads (overlaps with WMMA below)
    const int kn = k0 + KT;
    if (kn < K) {
#pragma unroll
      for (int i = 0; i < 4; i++) {
        int idx = tid + i * 256;
        aReg[i] = *(const float4*)(A + (size_t)(m0 + (idx >> 3)) * K + kn + (idx & 7) * 4);
      }
#pragma unroll
      for (int i = 0; i < 2; i++) {
        int idx = tid + i * 256;
        wReg[i] = *(const float4*)(W + (size_t)(kn + (idx >> 4)) * N + n0 + (idx & 15) * 4);
      }
    }

    // fragments: load all, then burst 4 WMMAs
    FragBF bf;
    bf.u[0] = *(const uint4*)&Wt[(nw + lm) * WT_STR + lh * 8];
    bf.u[1] = *(const uint4*)&Wt[(nw + lm) * WT_STR + lh * 8 + 16];
    FragBF af[4];
#pragma unroll
    for (int fm = 0; fm < 4; fm++) {
      int row = mw + fm * 16 + lm;
      af[fm].u[0] = *(const uint4*)&As[row * AS_STR + lh * 8];
      af[fm].u[1] = *(const uint4*)&As[row * AS_STR + lh * 8 + 16];
    }
#pragma unroll
    for (int fm = 0; fm < 4; fm++)
      acc[fm] = __builtin_amdgcn_wmma_f32_16x16x32_bf16(
          false, af[fm].v, false, bf.v, (short)0, acc[fm], false, false);
    __syncthreads();
  }

  const int n = n0 + nw + lm;
  const float bv = bias ? bias[n] : 0.0f;
#pragma unroll
  for (int fm = 0; fm < 4; fm++) {
#pragma unroll
    for (int r = 0; r < 8; r++) {
      int m = m0 + mw + fm * 16 + lh * 8 + r;
      C[(size_t)m * N + n] = acc[fm][r] + bv;
    }
  }
}

// ---------------------------------------------------------------------------
// LSH bucket argmax over concat(rot, -rot), first-max tie rule.
// ---------------------------------------------------------------------------
__global__ __launch_bounds__(256) void bucket_kernel(
    const float* __restrict__ rot, int* __restrict__ buckets) {
  int g = blockIdx.x * 256 + threadIdx.x;        // (b,h,t) flat
  int t = g & (TSEQ - 1);
  int h = (g >> 12) & (NH - 1);
  int b = g >> 15;
  const float* row = rot + ((size_t)(b * TSEQ + t)) * (NH * 32) + h * 32;
  float best = -INFINITY; int bi = 0;
#pragma unroll
  for (int i = 0; i < 32; i++) { float v = row[i];  if (v > best) { best = v; bi = i; } }
#pragma unroll
  for (int i = 0; i < 32; i++) { float v = -row[i]; if (v > best) { best = v; bi = i + 32; } }
  buckets[g] = bi;
}

// ---------------------------------------------------------------------------
// Stable counting sort per (b,h): 64 bins over 4096 tokens.
// st[j] = token at sorted pos j ; inv[t] = sorted pos of token t.
// ---------------------------------------------------------------------------
__global__ __launch_bounds__(64) void sort_kernel(
    const int* __restrict__ buckets, int* __restrict__ st, int* __restrict__ inv) {
  __shared__ int lb[TSEQ];
  __shared__ int hist[NB];
  __shared__ int offs[NB];
  const int bh = blockIdx.x;
  const int tid = threadIdx.x;
  const int* bk = buckets + (size_t)bh * TSEQ;
  hist[tid] = 0;
  __syncthreads();
  for (int t = tid; t < TSEQ; t += 64) {
    int c = bk[t];
    lb[t] = c;
    atomicAdd(&hist[c], 1);
  }
  __syncthreads();
  if (tid == 0) {
    int s = 0;
    for (int c = 0; c < NB; c++) { offs[c] = s; s += hist[c]; }
  }
  __syncthreads();
  int cnt = offs[tid];
  int* stp = st + (size_t)bh * TSEQ;
  int* ivp = inv + (size_t)bh * TSEQ;
  for (int t = 0; t < TSEQ; t++) {
    if (lb[t] == tid) { stp[cnt] = t; ivp[t] = cnt; cnt++; }
  }
}

// ---------------------------------------------------------------------------
// Chunk attention: gather 64 sorted rows, dots = bq*bq^T (col-scaled by
// 1/|bq_j| * D^-1/2), bo = dots * bv, write sorted output (bf16). All WMMA.
// ---------------------------------------------------------------------------
#define KQ_STR 520
#define VT_STR 72
#define DT_STR 72

__global__ __launch_bounds__(256) void attn_chunk_kernel(
    const float* __restrict__ qk, const float* __restrict__ v,
    const int* __restrict__ st, unsigned short* __restrict__ so) {
  __shared__ alignas(16) unsigned short Kq[BS * KQ_STR];     // bq bf16 [64][512]
  __shared__ alignas(16) unsigned short Vt[DDIM * VT_STR];   // bv^T bf16 [512][64]
  __shared__ alignas(16) unsigned short Dt[BS * DT_STR];     // dots bf16 [64][64]
  __shared__ float scl[BS];

  const int u = blockIdx.x;              // 0..511 (h*64 + chunk)
  const int b = blockIdx.y;              // batch
  const int h = u >> 6;
  const int jj0 = (u & 63) * BS;
  const int tid = threadIdx.x;
  const int wave = tid >> 5, lane = tid & 31;
  const int lm = lane & 15, lh = lane >> 4;

  if (tid < BS) scl[tid] = 0.0f;
  __syncthreads();

  { // gather + fp32->bf16 convert + row sumsq
    const int row = tid >> 2;
    const int seg = tid & 3;
    const int t = st[(b * NH + h) * TSEQ + jj0 + row];
    const float* qrow = qk + ((size_t)(b * TSEQ + t)) * DDIM;
    const float* vrow = v  + ((size_t)(b * TSEQ + t)) * DDIM;
    float ssq = 0.0f;
#pragma unroll
    for (int c4 = 0; c4 < 32; c4++) {
      const int c = seg * 128 + c4 * 4;
      float4 q = *(const float4*)(qrow + c);
      ssq += q.x * q.x + q.y * q.y + q.z * q.z + q.w * q.w;
      uint2 p; p.x = pack2(q.x, q.y); p.y = pack2(q.z, q.w);
      *(uint2*)&Kq[row * KQ_STR + c] = p;
      float4 w = *(const float4*)(vrow + c);
      Vt[(c + 0) * VT_STR + row] = f2bf(w.x);
      Vt[(c + 1) * VT_STR + row] = f2bf(w.y);
      Vt[(c + 2) * VT_STR + row] = f2bf(w.z);
      Vt[(c + 3) * VT_STR + row] = f2bf(w.w);
    }
    atomicAdd(&scl[row], ssq);
  }
  __syncthreads();
  if (tid < BS) {
    float nrm = fmaxf(sqrtf(scl[tid]), 1e-12f);
    scl[tid] = (1.0f / nrm) * 0.04419417382415922f;   // * D^-0.5
  }
  __syncthreads();

  { // dots GEMM 64x64, K=512
    const int rt = wave & 3;
    const int cp = wave >> 2;
    const int arow = rt * 16 + lm;
    const int n0r = (cp * 2 + 0) * 16 + lm;
    const int n1r = (cp * 2 + 1) * 16 + lm;
    v8f acc0 = {}, acc1 = {};
#pragma unroll
    for (int ks = 0; ks < 16; ks++) {
      const int k0 = ks * 32;
      FragBF af, b0, b1;
      af.u[0] = *(const uint4*)&Kq[arow * KQ_STR + k0 + lh * 8];
      af.u[1] = *(const uint4*)&Kq[arow * KQ_STR + k0 + lh * 8 + 16];
      b0.u[0] = *(const uint4*)&Kq[n0r * KQ_STR + k0 + lh * 8];
      b0.u[1] = *(const uint4*)&Kq[n0r * KQ_STR + k0 + lh * 8 + 16];
      b1.u[0] = *(const uint4*)&Kq[n1r * KQ_STR + k0 + lh * 8];
      b1.u[1] = *(const uint4*)&Kq[n1r * KQ_STR + k0 + lh * 8 + 16];
      acc0 = __builtin_amdgcn_wmma_f32_16x16x32_bf16(false, af.v, false, b0.v, (short)0, acc0, false, false);
      acc1 = __builtin_amdgcn_wmma_f32_16x16x32_bf16(false, af.v, false, b1.v, (short)0, acc1, false, false);
    }
#pragma unroll
    for (int e = 0; e < 2; e++) {
      const v8f a = e ? acc1 : acc0;
      const int n = (cp * 2 + e) * 16 + lm;
      const float s = scl[n];
#pragma unroll
      for (int r = 0; r < 8; r++) {
        const int m = rt * 16 + lh * 8 + r;
        Dt[m * DT_STR + n] = f2bf(a[r] * s);
      }
    }
  }
  __syncthreads();

  { // bo GEMM 64x512, K=64 ; write bf16 to sorted output
    const int rt = wave & 3;
    const int ch = wave >> 2;
    const int arow = rt * 16 + lm;
    FragBF a0, a1;
    a0.u[0] = *(const uint4*)&Dt[arow * DT_STR + lh * 8];
    a0.u[1] = *(const uint4*)&Dt[arow * DT_STR + lh * 8 + 16];
    a1.u[0] = *(const uint4*)&Dt[arow * DT_STR + 32 + lh * 8];
    a1.u[1] = *(const uint4*)&Dt[arow * DT_STR + 32 + lh * 8 + 16];
    unsigned short* sob = so + (((size_t)b * (NH * TSEQ)) + (size_t)h * TSEQ + jj0) * DDIM;
#pragma unroll
    for (int dt = 0; dt < 16; dt++) {
      const int d = ch * 256 + dt * 16 + lm;
      FragBF b0, b1;
      b0.u[0] = *(const uint4*)&Vt[d * VT_STR + lh * 8];
      b0.u[1] = *(const uint4*)&Vt[d * VT_STR + lh * 8 + 16];
      b1.u[0] = *(const uint4*)&Vt[d * VT_STR + 32 + lh * 8];
      b1.u[1] = *(const uint4*)&Vt[d * VT_STR + 32 + lh * 8 + 16];
      v8f acc = {};
      acc = __builtin_amdgcn_wmma_f32_16x16x32_bf16(false, a0.v, false, b0.v, (short)0, acc, false, false);
      acc = __builtin_amdgcn_wmma_f32_16x16x32_bf16(false, a1.v, false, b1.v, (short)0, acc, false, false);
#pragma unroll
      for (int r = 0; r < 8; r++) {
        const int m = rt * 16 + lh * 8 + r;
        sob[(size_t)m * DDIM + d] = f2bf(acc[r]);
      }
    }
  }
}

// ---------------------------------------------------------------------------
// o[b,t,:] = sum_h so_bf16[b, h*T + inv[b,h,t], :]   (fp32 accumulate)
// ---------------------------------------------------------------------------
__global__ __launch_bounds__(128) void combine_kernel(
    const unsigned short* __restrict__ so, const int* __restrict__ inv,
    float* __restrict__ o) {
  const int bt = blockIdx.x;
  const int b = bt >> 12;
  const int t = bt & (TSEQ - 1);
  const int d = threadIdx.x * 4;
  float4 acc = make_float4(0.f, 0.f, 0.f, 0.f);
#pragma unroll
  for (int h = 0; h < NH; h++) {
    int j = inv[(b * NH + h) * TSEQ + t];
    uint2 p = *(const uint2*)(so + (((size_t)b * (NH * TSEQ)) + (size_t)h * TSEQ + j) * DDIM + d);
    acc.x += __uint_as_float(p.x << 16);
    acc.y += __uint_as_float(p.x & 0xffff0000u);
    acc.z += __uint_as_float(p.y << 16);
    acc.w += __uint_as_float(p.y & 0xffff0000u);
  }
  *(float4*)(o + (size_t)bt * DDIM + d) = acc;
}

// ---------------------------------------------------------------------------
extern "C" void kernel_launch(void* const* d_in, const int* in_sizes, int n_in,
                              void* d_out, int out_size, void* d_ws, size_t ws_size,
                              hipStream_t stream) {
  const float* x     = (const float*)d_in[0];
  const float* rotw  = (const float*)d_in[1];   // [512][8][32] == [512][256]
  const float* w_qk  = (const float*)d_in[2];
  const float* b_qk  = (const float*)d_in[3];
  const float* w_v   = (const float*)d_in[4];
  const float* b_v   = (const float*)d_in[5];
  const float* w_out = (const float*)d_in[6];
  const float* b_out = (const float*)d_in[7];

  const int B = 4, T = TSEQ, D = DDIM, M = B * T;   // M = 16384

  char* p = (char*)d_ws;
  auto alloc = [&](size_t bytes) {
    char* r = p;
    p += (bytes + 255) & ~(size_t)255;
    return r;
  };
  float* qk   = (float*)alloc((size_t)M * D * 4);            // 32 MB
  float* vv   = (float*)alloc((size_t)M * D * 4);            // 32 MB
  float* rot  = (float*)alloc((size_t)M * (NH * 32) * 4);    // 16 MB
  int*   bkt  = (int*)  alloc((size_t)B * NH * T * 4);
  int*   st   = (int*)  alloc((size_t)B * NH * T * 4);
  int*   inv  = (int*)  alloc((size_t)B * NH * T * 4);
  unsigned short* so = (unsigned short*)alloc((size_t)B * NH * T * D * 2);  // 128 MB bf16
  float* osum = (float*)alloc((size_t)M * D * 4);            // 32 MB
  (void)ws_size; (void)in_sizes; (void)n_in; (void)out_size;

  const dim3 blk(256);
  gemm_bf16<<<dim3(M / BM, D / BN), blk, 0, stream>>>(x, w_qk, b_qk, qk, M, D, D);
  gemm_bf16<<<dim3(M / BM, D / BN), blk, 0, stream>>>(x, w_v, b_v, vv, M, D, D);
  gemm_bf16<<<dim3(M / BM, (NH * 32) / BN), blk, 0, stream>>>(qk, rotw, nullptr, rot, M, NH * 32, D);
  bucket_kernel<<<dim3((B * NH * T) / 256), blk, 0, stream>>>(rot, bkt);
  sort_kernel<<<dim3(B * NH), dim3(64), 0, stream>>>(bkt, st, inv);
  attn_chunk_kernel<<<dim3(NH * NB, B), blk, 0, stream>>>(qk, vv, st, so);
  combine_kernel<<<dim3(M), dim3(128), 0, stream>>>(so, inv, osum);
  gemm_bf16<<<dim3(M / BM, D / BN), blk, 0, stream>>>(osum, w_out, b_out, (float*)d_out, M, D, D);
}
